// CriticNetwork_67508295958578
// MI455X (gfx1250) — compile-verified, matrix operand order checked by reading, not run on hardware
//
#include <hip/hip_runtime.h>
#include <hip/hip_bf16.h>

#define BATCH 32768
#define DIN   128
#define ACTP  32      // action dim padded 16 -> 32
#define NH1   600
#define NH1P  608     // padded to 38*16
#define NH1S  616     // LDS row stride (halves): conflict-free b128 fragment reads
#define NH2   500
#define NH2P  512     // padded to 32*16
#define NGAME 8
#define ROWS  32      // rows per block
#define NT1   38      // layer-1 n-tiles (608/16)
#define KT2   19      // layer-2 k-steps (608/32)

typedef __bf16 bf16x16 __attribute__((ext_vector_type(16)));
typedef __bf16 bf16x8  __attribute__((ext_vector_type(8)));
typedef float  f32x8   __attribute__((ext_vector_type(8)));

union BFrag { bf16x16 v; bf16x8 h[2]; };

__device__ __forceinline__ void load_bfrag(BFrag& f, const __bf16* p) {
  f.h[0] = *(const bf16x8*)p;
  f.h[1] = *(const bf16x8*)(p + 16);
}

// ---------------- prep: f32 -> bf16 with zero padding ----------------
__global__ void k_cvt_state(const float* __restrict__ s, __bf16* __restrict__ o) {
  int i = blockIdx.x * 256 + threadIdx.x;
  if (i < BATCH * DIN) o[i] = (__bf16)s[i];
}
__global__ void k_cvt_action(const float* __restrict__ a, __bf16* __restrict__ o) {
  int i = blockIdx.x * 256 + threadIdx.x;
  if (i < BATCH * ACTP) {
    int b = i >> 5, c = i & 31;
    o[i] = (c < 16) ? (__bf16)a[b * 16 + c] : (__bf16)0.0f;
  }
}
__global__ void k_cvt_w1(const float* __restrict__ w, __bf16* __restrict__ o) {
  int i = blockIdx.x * 256 + threadIdx.x;
  if (i < NGAME * NH1P * DIN) {
    int g = i / (NH1P * DIN);
    int r = i % (NH1P * DIN);
    int n = r / DIN, k = r % DIN;
    o[i] = (n < NH1) ? (__bf16)w[(g * NH1 + n) * DIN + k] : (__bf16)0.0f;
  }
}
__global__ void k_cvt_w2s(const float* __restrict__ w, __bf16* __restrict__ o) {
  int i = blockIdx.x * 256 + threadIdx.x;
  if (i < NH2P * NH1P) {
    int n = i / NH1P, k = i % NH1P;
    o[i] = (n < NH2 && k < NH1) ? (__bf16)w[n * NH1 + k] : (__bf16)0.0f;
  }
}
__global__ void k_cvt_w2a(const float* __restrict__ w, __bf16* __restrict__ o) {
  int i = blockIdx.x * 256 + threadIdx.x;
  if (i < NGAME * NH2P * ACTP) {
    int g = i / (NH2P * ACTP);
    int r = i % (NH2P * ACTP);
    int n = r / ACTP, a2 = r % ACTP;
    o[i] = (n < NH2 && a2 < 16) ? (__bf16)w[(g * NH2 + n) * 16 + a2] : (__bf16)0.0f;
  }
}

#define WMMA_BF16(A, Bv, C) \
  __builtin_amdgcn_wmma_f32_16x16x32_bf16(false, (A), false, (Bv), (short)0, (C), false, false)

// ---------------- fused critic forward ----------------
__global__ __launch_bounds__(256) void k_critic_fused(
    const __bf16* __restrict__ stateB, const __bf16* __restrict__ actB,
    const __bf16* __restrict__ w1B,   const __bf16* __restrict__ w2sB,
    const __bf16* __restrict__ w2aB,
    const float* __restrict__ b1,  const float* __restrict__ b2s,
    const float* __restrict__ W3,  const float* __restrict__ b3,
    const int* __restrict__ idx,   float* __restrict__ qout)
{
  __shared__ __bf16 h1s[ROWS * NH1S];   // bf16 h1 tile, padded stride
  __shared__ float  qps[8][ROWS];       // per-wave q partials
  __shared__ int    idx_s[ROWS];

  const int tid   = threadIdx.x;
  const int lane  = tid & 31;
  const int wave  = tid >> 5;           // 0..7
  const int rlane = lane & 15;
  const int hi    = (lane >= 16) ? 1 : 0;
  const int koff  = hi * 8;             // K offset for 16-bit A/B fragment halves
  const int hi8   = hi * 8;             // C-fragment row offset
  const int r0    = blockIdx.x * ROWS;

  if (tid < ROWS) idx_s[tid] = idx[r0 + tid];
  __syncthreads();
  // block-uniform game range -> SGPRs (scalar branch / scalar weight bases)
  const int gLo = __builtin_amdgcn_readfirstlane(idx_s[0]);
  const int gHi = __builtin_amdgcn_readfirstlane(idx_s[ROWS - 1]);   // idx sorted
  const bool oneGame = (gLo == gHi);    // true for >= 1017/1024 blocks

  // ============ layer 1: h1 = relu(state @ W1[g]^T + b1[g]) -> LDS ============
  BFrag a1[2][4];
  #pragma unroll
  for (int mm = 0; mm < 2; ++mm) {
    const __bf16* sp = stateB + (size_t)(r0 + mm * 16 + rlane) * DIN + koff;
    #pragma unroll
    for (int k = 0; k < 4; ++k) load_bfrag(a1[mm][k], sp + k * 32);
  }

  const int ntile1 = (NT1 - wave * 5) < 5 ? (NT1 - wave * 5) : 5;  // 5, wave7: 3
  const int vb1    = (wave * 5 * 16 + rlane) * DIN + koff;         // per-lane B offset

  if (oneGame) {
    // ---- fast path: single routed game, software-pipelined over n-tiles ----
    const __bf16* w1g = w1B + (size_t)gLo * (NH1P * DIN);          // scalar base
    const float*  b1g = b1 + gLo * NH1;

    BFrag bP[4], bQ[4];
    #define L1_LOAD(BV, tt)                                              \
      do {                                                               \
        _Pragma("unroll")                                                \
        for (int k_ = 0; k_ < 4; ++k_)                                   \
          load_bfrag((BV)[k_], w1g + vb1 + (tt) * (16 * DIN) + k_ * 32); \
      } while (0)
    #define L1_STEP(BV, tt)                                              \
      do {                                                               \
        f32x8 c0 = {0.f,0.f,0.f,0.f,0.f,0.f,0.f,0.f};                    \
        f32x8 c1 = c0;                                                   \
        _Pragma("unroll")                                                \
        for (int k_ = 0; k_ < 4; ++k_) {                                 \
          c0 = WMMA_BF16(a1[0][k_].v, (BV)[k_].v, c0);                   \
          c1 = WMMA_BF16(a1[1][k_].v, (BV)[k_].v, c1);                   \
        }                                                                \
        const int ng_ = (wave * 5 + (tt)) * 16 + rlane;                  \
        const float bias_ = (ng_ < NH1) ? b1g[ng_] : 0.f;                \
        _Pragma("unroll")                                                \
        for (int j_ = 0; j_ < 8; ++j_) {                                 \
          float v0 = c0[j_] + bias_; v0 = v0 > 0.f ? v0 : 0.f;           \
          float v1 = c1[j_] + bias_; v1 = v1 > 0.f ? v1 : 0.f;           \
          h1s[(hi8 + j_) * NH1S + ng_]      = (__bf16)v0;                \
          h1s[(16 + hi8 + j_) * NH1S + ng_] = (__bf16)v1;                \
        }                                                                \
      } while (0)

    L1_LOAD(bP, 0);
    for (int t = 0; ; t += 2) {
      if (t + 1 < ntile1) L1_LOAD(bQ, t + 1);   // prefetch t+1
      L1_STEP(bP, t);
      if (t + 1 >= ntile1) break;
      if (t + 2 < ntile1) L1_LOAD(bP, t + 2);   // prefetch t+2
      L1_STEP(bQ, t + 1);
      if (t + 2 >= ntile1) break;
    }
    #undef L1_LOAD
    #undef L1_STEP
  } else {
    // ---- generic path (<= 7 boundary blocks): select-accumulate over games ----
    int gi[2][8];
    #pragma unroll
    for (int mm = 0; mm < 2; ++mm)
      #pragma unroll
      for (int j = 0; j < 8; ++j) gi[mm][j] = idx_s[mm * 16 + hi8 + j];

    for (int t = 0; t < ntile1; ++t) {
      const int n  = wave * 5 + t;
      const int ng = n * 16 + rlane;
      float o0[8], o1[8];
      #pragma unroll
      for (int j = 0; j < 8; ++j) { o0[j] = 0.f; o1[j] = 0.f; }

      for (int g = gLo; g <= gHi; ++g) {
        const __bf16* w1g = w1B + (size_t)g * (NH1P * DIN);   // scalar base
        BFrag b[4];
        #pragma unroll
        for (int k = 0; k < 4; ++k)
          load_bfrag(b[k], w1g + vb1 + t * (16 * DIN) + k * 32);
        f32x8 c0 = {0.f,0.f,0.f,0.f,0.f,0.f,0.f,0.f};
        f32x8 c1 = c0;
        #pragma unroll
        for (int k = 0; k < 4; ++k) {
          c0 = WMMA_BF16(a1[0][k].v, b[k].v, c0);
          c1 = WMMA_BF16(a1[1][k].v, b[k].v, c1);
        }
        const float bias = (ng < NH1) ? b1[g * NH1 + ng] : 0.f;
        #pragma unroll
        for (int j = 0; j < 8; ++j) {             // branchless routed select
          float v0 = c0[j] + bias; v0 = v0 > 0.f ? v0 : 0.f;
          float v1 = c1[j] + bias; v1 = v1 > 0.f ? v1 : 0.f;
          o0[j] = (gi[0][j] == g) ? v0 : o0[j];
          o1[j] = (gi[1][j] == g) ? v1 : o1[j];
        }
      }
      #pragma unroll
      for (int j = 0; j < 8; ++j) {
        h1s[(hi8 + j) * NH1S + ng]      = (__bf16)o0[j];
        h1s[(16 + hi8 + j) * NH1S + ng] = (__bf16)o1[j];
      }
    }
  }
  __syncthreads();

  // ============ layer 2: h2 = h1 @ W2s^T + b2s + action @ W2a[g]^T ============
  BFrag aa[2];
  #pragma unroll
  for (int mm = 0; mm < 2; ++mm)
    load_bfrag(aa[mm], actB + (size_t)(r0 + mm * 16 + rlane) * ACTP + koff);

  const int aOff0 = rlane * NH1S + koff;            // LDS offsets (elements)
  const int aOff1 = (16 + rlane) * NH1S + koff;
  const int bOff  = ((wave * 4) * 16 + rlane) * NH1P + koff;  // 32-bit voffset

  f32x8 acc[2][4];
  {
    f32x8 z = {0.f,0.f,0.f,0.f,0.f,0.f,0.f,0.f};
    #pragma unroll
    for (int mm = 0; mm < 2; ++mm)
      #pragma unroll
      for (int i = 0; i < 4; ++i) acc[mm][i] = z;
  }

  // software-pipelined K loop: double-buffered A (LDS) + B (global) fragments
  BFrag aP[2], bP2[4], aQ[2], bQ2[4];
  #define L2_LOAD(AV, BV, kk)                                           \
    do {                                                                \
      load_bfrag((AV)[0], &h1s[aOff0 + (kk) * 32]);                     \
      load_bfrag((AV)[1], &h1s[aOff1 + (kk) * 32]);                     \
      _Pragma("unroll")                                                 \
      for (int i_ = 0; i_ < 4; ++i_)                                    \
        load_bfrag((BV)[i_], w2sB + bOff + i_ * (16 * NH1P) + (kk) * 32); \
    } while (0)
  #define L2_MMA(AV, BV)                                                \
    do {                                                                \
      _Pragma("unroll")                                                 \
      for (int i_ = 0; i_ < 4; ++i_) {                                  \
        acc[0][i_] = WMMA_BF16((AV)[0].v, (BV)[i_].v, acc[0][i_]);      \
        acc[1][i_] = WMMA_BF16((AV)[1].v, (BV)[i_].v, acc[1][i_]);      \
      }                                                                 \
    } while (0)

  L2_LOAD(aP, bP2, 0);
  for (int k = 0; ; k += 2) {
    if (k + 1 < KT2) L2_LOAD(aQ, bQ2, k + 1);   // prefetch k+1
    L2_MMA(aP, bP2);                             // compute k
    if (k + 1 >= KT2) break;
    if (k + 2 < KT2) L2_LOAD(aP, bP2, k + 2);   // prefetch k+2
    L2_MMA(aQ, bQ2);                             // compute k+1
    if (k + 2 >= KT2) break;
  }
  #undef L2_LOAD
  #undef L2_MMA

  float qp[2][8];
  #pragma unroll
  for (int mm = 0; mm < 2; ++mm)
    #pragma unroll
    for (int j = 0; j < 8; ++j) qp[mm][j] = 0.f;

  if (oneGame) {
    // ---- fast path epilogue: uniform game, no routing selects ----
    const __bf16* w2ag = w2aB + (size_t)gLo * (NH2P * ACTP);   // scalar base
    const float*  W3g  = W3 + gLo * NH2;
    BFrag bw[4];
    #pragma unroll
    for (int i = 0; i < 4; ++i)
      load_bfrag(bw[i], w2ag + ((wave * 4 + i) * 16 + rlane) * ACTP + koff);
    #pragma unroll
    for (int i = 0; i < 4; ++i) {
      const int   ng = (wave * 4 + i) * 16 + rlane;
      const float b2 = (ng < NH2) ? b2s[ng] : 0.f;
      const float w3 = (ng < NH2) ? W3g[ng] : 0.f;
      f32x8 z = {0.f,0.f,0.f,0.f,0.f,0.f,0.f,0.f};
      f32x8 d0 = WMMA_BF16(aa[0].v, bw[i].v, z);
      f32x8 d1 = WMMA_BF16(aa[1].v, bw[i].v, z);
      #pragma unroll
      for (int j = 0; j < 8; ++j) {
        float h0 = acc[0][i][j] + b2 + d0[j]; h0 = h0 > 0.f ? h0 : 0.f;
        float h1 = acc[1][i][j] + b2 + d1[j]; h1 = h1 > 0.f ? h1 : 0.f;
        qp[0][j] += h0 * w3;
        qp[1][j] += h1 * w3;
      }
    }
  } else {
    // ---- generic epilogue: per-row routed action term + W3 ----
    int gi[2][8];
    #pragma unroll
    for (int mm = 0; mm < 2; ++mm)
      #pragma unroll
      for (int j = 0; j < 8; ++j) gi[mm][j] = idx_s[mm * 16 + hi8 + j];

    #pragma unroll
    for (int i = 0; i < 4; ++i) {
      const int   n  = wave * 4 + i;
      const int   ng = n * 16 + rlane;
      const float b2 = (ng < NH2) ? b2s[ng] : 0.f;
      float hv[2][8];
      #pragma unroll
      for (int mm = 0; mm < 2; ++mm)
        #pragma unroll
        for (int j = 0; j < 8; ++j) hv[mm][j] = acc[mm][i][j] + b2;

      for (int g = gLo; g <= gHi; ++g) {
        const __bf16* w2ag = w2aB + (size_t)g * (NH2P * ACTP);  // scalar base
        BFrag bw;
        load_bfrag(bw, w2ag + (n * 16 + rlane) * ACTP + koff);
        f32x8 z = {0.f,0.f,0.f,0.f,0.f,0.f,0.f,0.f};
        f32x8 d0 = WMMA_BF16(aa[0].v, bw.v, z);
        f32x8 d1 = WMMA_BF16(aa[1].v, bw.v, z);
        #pragma unroll
        for (int j = 0; j < 8; ++j) {            // branchless: v_cndmask + add
          hv[0][j] += (gi[0][j] == g) ? d0[j] : 0.f;
          hv[1][j] += (gi[1][j] == g) ? d1[j] : 0.f;
        }
      }
      #pragma unroll
      for (int mm = 0; mm < 2; ++mm)
        #pragma unroll
        for (int j = 0; j < 8; ++j) {
          float h = hv[mm][j] > 0.f ? hv[mm][j] : 0.f;   // relu
          const float w3v = (ng < NH2) ? W3[gi[mm][j] * NH2 + ng] : 0.f;
          qp[mm][j] += h * w3v;
        }
    }
  }

  // reduce over the 16 column lanes (xor masks < 16 stay within each half)
  #pragma unroll
  for (int mm = 0; mm < 2; ++mm)
    #pragma unroll
    for (int j = 0; j < 8; ++j) {
      float v = qp[mm][j];
      v += __shfl_xor(v, 1, 32);
      v += __shfl_xor(v, 2, 32);
      v += __shfl_xor(v, 4, 32);
      v += __shfl_xor(v, 8, 32);
      if (rlane == 0) qps[wave][mm * 16 + hi8 + j] = v;
    }
  __syncthreads();

  if (tid < ROWS) {
    float s = 0.f;
    #pragma unroll
    for (int w = 0; w < 8; ++w) s += qps[w][tid];   // fixed order: deterministic
    qout[r0 + tid] = s + b3[idx_s[tid]];
  }
}

extern "C" void kernel_launch(void* const* d_in, const int* in_sizes, int n_in,
                              void* d_out, int out_size, void* d_ws, size_t ws_size,
                              hipStream_t stream) {
  (void)in_sizes; (void)n_in; (void)out_size; (void)ws_size;
  const float* state  = (const float*)d_in[0];
  const float* action = (const float*)d_in[1];
  const int*   idx    = (const int*)d_in[2];
  const float* W1     = (const float*)d_in[3];
  const float* b1     = (const float*)d_in[4];
  const float* W2s    = (const float*)d_in[5];
  const float* b2s    = (const float*)d_in[6];
  const float* W2a    = (const float*)d_in[7];
  const float* W3     = (const float*)d_in[8];
  const float* b3     = (const float*)d_in[9];
  float* qout = (float*)d_out;

  char* p = (char*)d_ws;
  __bf16* stateB = (__bf16*)p;  p += (size_t)BATCH * DIN  * 2;
  __bf16* actB   = (__bf16*)p;  p += (size_t)BATCH * ACTP * 2;
  __bf16* w1B    = (__bf16*)p;  p += (size_t)NGAME * NH1P * DIN  * 2;
  __bf16* w2sB   = (__bf16*)p;  p += (size_t)NH2P  * NH1P * 2;
  __bf16* w2aB   = (__bf16*)p;  p += (size_t)NGAME * NH2P * ACTP * 2;

  k_cvt_state <<<(BATCH * DIN        + 255) / 256, 256, 0, stream>>>(state,  stateB);
  k_cvt_action<<<(BATCH * ACTP       + 255) / 256, 256, 0, stream>>>(action, actB);
  k_cvt_w1    <<<(NGAME * NH1P * DIN + 255) / 256, 256, 0, stream>>>(W1,  w1B);
  k_cvt_w2s   <<<(NH2P * NH1P        + 255) / 256, 256, 0, stream>>>(W2s, w2sB);
  k_cvt_w2a   <<<(NGAME * NH2P * ACTP+ 255) / 256, 256, 0, stream>>>(W2a, w2aB);

  k_critic_fused<<<BATCH / ROWS, 256, 0, stream>>>(stateB, actB, w1B, w2sB, w2aB,
                                                   b1, b2s, W3, b3, idx, qout);
}
